// CausalSelfAttention_29987461661368
// MI455X (gfx1250) — compile-verified, matrix operand order checked by reading, not run on hardware
//
#include <hip/hip_runtime.h>
#include <stdint.h>

// ---------------------------------------------------------------------------
// Problem constants (from reference): B=2, S=2048, H=16, D=128
// ---------------------------------------------------------------------------
#define B_    2
#define S_    2048
#define H_    16
#define D_    128
#define HID_  (H_ * D_)           // 2048
#define M_    (B_ * S_)           // 4096 rows of X
#define SCALE_ 0.08838834764831845f   // 1/sqrt(128)

// ---------------------------------------------------------------------------
// CDNA5 WMMA types (wave32): A/B are 16 bf16 per lane, C/D are 8 f32 per lane
// ---------------------------------------------------------------------------
typedef __attribute__((ext_vector_type(16))) __bf16 v16bf;
typedef __attribute__((ext_vector_type(8)))  __bf16 v8bf;
typedef __attribute__((ext_vector_type(8)))  float  v8f;

__device__ __forceinline__ unsigned short f2bf(float f) {
    union { float f; unsigned int u; } c; c.f = f;
    unsigned int u = c.u;
    unsigned int r = (u + 0x7FFFu + ((u >> 16) & 1u)) >> 16;  // RNE
    return (unsigned short)r;
}
__device__ __forceinline__ float bf2f(unsigned short h) {
    union { float f; unsigned int u; } c; c.u = ((unsigned int)h) << 16;
    return c.f;
}
__device__ __forceinline__ v16bf concatbf(v8bf lo, v8bf hi) {
    return __builtin_shufflevector(lo, hi, 0,1,2,3,4,5,6,7,8,9,10,11,12,13,14,15);
}
__device__ __forceinline__ v8f wmma_bf16(v16bf a, v16bf b, v8f c) {
    // D(16x16,f32) = A(16x32,bf16) x B(32x16,bf16) + C
    return __builtin_amdgcn_wmma_f32_16x16x32_bf16(
        /*neg_a=*/false, a, /*neg_b=*/false, b,
        /*c_mod=*/(short)0, c, /*reuse_a=*/false, /*reuse_b=*/false);
}

// LDS fragment loader: ISA 16-bit 16x32 A/B layout.
// lanes 0-15: row = lane,    K chunks {c..c+7} and {c+16..c+23}, c=0
// lanes16-31: row = lane-16, same with c=8
__device__ __forceinline__ v16bf frag_from_lds(const unsigned short* base, int row, int g8) {
    v8bf lo = *(const v8bf*)(base + row * 32 + g8);
    v8bf hi = *(const v8bf*)(base + row * 32 + g8 + 16);
    return concatbf(lo, hi);
}

// ---------------------------------------------------------------------------
// Cooperative 128x32-element tile: global -> regs, then regs -> LDS (as bf16).
// 256 threads, 4 chunks of 4 elements each.
// ---------------------------------------------------------------------------
template<typename T>
__device__ __forceinline__ void gload_tile(const T* __restrict__ src, int ld, int tid,
                                           uint4* __restrict__ regs) {
#pragma unroll
    for (int it = 0; it < 4; ++it) {
        int linear = tid + 256 * it;          // 0..1023
        int row = linear >> 3;                // 0..127
        int c4  = (linear & 7) * 4;           // 0..28
        if constexpr (sizeof(T) == 4) {
            regs[it] = *(const uint4*)(src + (size_t)row * ld + c4);   // 4 raw f32
        } else {
            uint2 v = *(const uint2*)(src + (size_t)row * ld + c4);    // 4 bf16
            regs[it].x = v.x; regs[it].y = v.y;
        }
    }
}

template<typename T>
__device__ __forceinline__ void stage_tile(const uint4* __restrict__ regs,
                                           unsigned short* __restrict__ dst, int tid) {
#pragma unroll
    for (int it = 0; it < 4; ++it) {
        int linear = tid + 256 * it;
        int row = linear >> 3;
        int c4  = (linear & 7) * 4;
        if constexpr (sizeof(T) == 4) {
            float4 v = *(const float4*)&regs[it];
            unsigned int p0 = ((unsigned int)f2bf(v.y) << 16) | (unsigned int)f2bf(v.x);
            unsigned int p1 = ((unsigned int)f2bf(v.w) << 16) | (unsigned int)f2bf(v.z);
            *(uint2*)(dst + row * 32 + c4) = make_uint2(p0, p1);
        } else {
            *(uint2*)(dst + row * 32 + c4) = make_uint2(regs[it].x, regs[it].y);
        }
    }
}

// ---------------------------------------------------------------------------
// GEMM: O(MxN) = A(MxK) * W(NxK)^T   (W rows contiguous => direct B-columns)
// Block tile 128x128, 8 waves (2x4), each wave 64x32 = 8 WMMA tiles per step.
// Software-pipelined: double-buffered LDS, register prefetch of next K-tile,
// one barrier per K-step; global-load latency hides behind the WMMAs.
// ---------------------------------------------------------------------------
template<typename TA, typename TB, typename TOUT>
__global__ __launch_bounds__(256) void gemm_xwt(const TA* __restrict__ Xm,
                                                const TB* __restrict__ Wm,
                                                TOUT* __restrict__ Om,
                                                int M, int N, int K) {
    __shared__ __align__(16) unsigned short sA[2][128 * 32];
    __shared__ __align__(16) unsigned short sB[2][128 * 32];

    const int tid     = threadIdx.x;
    const int lane    = tid & 31;
    const int wave    = tid >> 5;
    const int wm      = wave >> 2;         // 0..1  -> 64 rows each
    const int wn      = wave & 3;          // 0..3  -> 32 cols each
    const int gm0     = blockIdx.y * 128;
    const int gn0     = blockIdx.x * 128;
    const int laneRow = lane & 15;
    const int g8      = (lane >> 4) * 8;

    const TA* Abase = Xm + (size_t)gm0 * K;
    const TB* Bbase = Wm + (size_t)gn0 * K;

    v8f acc[4][2] = {};
    uint4 ra[4], rb[4];

    // prologue: tile 0 -> regs -> LDS[0]
    gload_tile<TA>(Abase, K, tid, ra);
    gload_tile<TB>(Bbase, K, tid, rb);
    stage_tile<TA>(ra, sA[0], tid);
    stage_tile<TB>(rb, sB[0], tid);
    __syncthreads();

    const int nIter = K / 32;
    for (int i = 0; i < nIter; ++i) {
        const int p = i & 1;
        // prefetch next K-tile into registers (latency overlapped with WMMAs)
        if (i + 1 < nIter) {
            gload_tile<TA>(Abase + (i + 1) * 32, K, tid, ra);
            gload_tile<TB>(Bbase + (i + 1) * 32, K, tid, rb);
        }
        // consume LDS buffer p
        v16bf af[4], bfr[2];
#pragma unroll
        for (int ii = 0; ii < 4; ++ii)
            af[ii] = frag_from_lds(sA[p], wm * 64 + ii * 16 + laneRow, g8);
#pragma unroll
        for (int j = 0; j < 2; ++j)
            bfr[j] = frag_from_lds(sB[p], wn * 32 + j * 16 + laneRow, g8);
#pragma unroll
        for (int ii = 0; ii < 4; ++ii)
#pragma unroll
            for (int j = 0; j < 2; ++j)
                acc[ii][j] = wmma_bf16(af[ii], bfr[j], acc[ii][j]);
        // stage next tile into the other buffer (safe: consumed at iter i-1)
        if (i + 1 < nIter) {
            stage_tile<TA>(ra, sA[1 - p], tid);
            stage_tile<TB>(rb, sB[1 - p], tid);
        }
        __syncthreads();
    }

    // C layout: VGPR r -> (M=r, lanes0-15) and (M=r+8, lanes16-31), N=lane&15
#pragma unroll
    for (int ii = 0; ii < 4; ++ii)
#pragma unroll
        for (int j = 0; j < 2; ++j)
#pragma unroll
            for (int r = 0; r < 8; ++r) {
                int m = gm0 + wm * 64 + ii * 16 + r + ((lane >> 4) << 3);
                int n = gn0 + wn * 32 + j * 16 + laneRow;
                float v = acc[ii][j][r];
                if constexpr (sizeof(TOUT) == 4) Om[(size_t)m * N + n] = v;
                else                             Om[(size_t)m * N + n] = f2bf(v);
            }
}

// ---------------------------------------------------------------------------
// RoPE + split: qkv(B,S,3,H,D) bf16 -> q,k (roped), v  in (B,H,S,D) bf16
// ---------------------------------------------------------------------------
__global__ __launch_bounds__(256) void rope_split(const unsigned short* __restrict__ qkv,
                                                  unsigned short* __restrict__ q,
                                                  unsigned short* __restrict__ k,
                                                  unsigned short* __restrict__ v) {
    int gid = blockIdx.x * 256 + threadIdx.x;     // B*H*S*(D/2) threads
    int d = gid & 63;
    int s = (gid >> 6) & (S_ - 1);
    int h = (gid >> 17) & (H_ - 1);
    int b = gid >> 21;

    size_t inRow = (size_t)(b * S_ + s) * (3 * HID_);
    float q0 = bf2f(qkv[inRow + h * D_ + d]);
    float q1 = bf2f(qkv[inRow + h * D_ + d + 64]);
    float k0 = bf2f(qkv[inRow + (H_ + h) * D_ + d]);
    float k1 = bf2f(qkv[inRow + (H_ + h) * D_ + d + 64]);

    // inv_freq = theta^(-2d/D) ; angle = s * inv_freq
    float invf = __expf(-((float)(2 * d) / (float)D_) * 9.210340371976184f);
    float ang  = (float)s * invf;
    float c = __cosf(ang), sn = __sinf(ang);

    size_t outRow = (size_t)((b * H_ + h) * S_ + s) * D_;
    q[outRow + d]      = f2bf(q0 * c - q1 * sn);
    q[outRow + d + 64] = f2bf(q1 * c + q0 * sn);
    k[outRow + d]      = f2bf(k0 * c - k1 * sn);
    k[outRow + d + 64] = f2bf(k1 * c + k0 * sn);
    v[outRow + d]      = qkv[inRow + (2 * H_ + h) * D_ + d];
    v[outRow + d + 64] = qkv[inRow + (2 * H_ + h) * D_ + d + 64];
}

// ---------------------------------------------------------------------------
// Flash attention (causal): block = (128 q-rows of one (b,h)), 8 waves x16 rows.
// Pipelined: stage tile t -> barrier -> (prefetch tile t+1 || compute t) -> barrier.
// Per processed tile: scores = Q·K^T (8 WMMA), online softmax, O += P·V (8 WMMA).
// ---------------------------------------------------------------------------
__global__ __launch_bounds__(256) void flash_attn(const unsigned short* __restrict__ Q,
                                                  const unsigned short* __restrict__ Kb,
                                                  const unsigned short* __restrict__ Vb,
                                                  unsigned short* __restrict__ Ob) {
    __shared__ __align__(16) unsigned short sK[32 * 128];   // (key, d)
    __shared__ __align__(16) unsigned short sV[128 * 32];   // (d, key)  transposed
    __shared__ __align__(16) unsigned short sP[8 * 16 * 32];// per-wave P tile

    const int tid     = threadIdx.x;
    const int lane    = tid & 31;
    const int wave    = tid >> 5;
    const int laneRow = lane & 15;
    const int g8      = (lane >> 4) * 8;
    const int hiOff   = (lane >> 4) << 3;

    const int bh = blockIdx.y;               // b*H + h
    const int q0 = blockIdx.x * 128;
    const int qw = q0 + wave * 16;           // this wave's 16 query rows
    const size_t baseBH = (size_t)bh * S_ * D_;

    // Persistent Q fragments: 16 rows x 128 dims = 4 K-slices of 32
    v16bf qf[4];
    {
        const unsigned short* qrow = Q + baseBH + (size_t)(qw + laneRow) * D_;
#pragma unroll
        for (int s4 = 0; s4 < 4; ++s4) {
            int c = s4 * 32 + g8;
            qf[s4] = concatbf(*(const v8bf*)(qrow + c), *(const v8bf*)(qrow + c + 16));
        }
    }

    v8f oacc[8] = {};
    float mi[8], li[8];
#pragma unroll
    for (int r = 0; r < 8; ++r) { mi[r] = -3.0e38f; li[r] = 0.0f; }

    uint4 kreg[2], vreg[2];
    auto prefetch = [&](int k0) {
#pragma unroll
        for (int it = 0; it < 2; ++it) {
            int idx = tid + 256 * it;            // 0..511
            int row = idx >> 4;                  // key 0..31
            int c8  = (idx & 15) * 8;            // d 0..120
            kreg[it] = *(const uint4*)(Kb + baseBH + (size_t)(k0 + row) * D_ + c8);
            vreg[it] = *(const uint4*)(Vb + baseBH + (size_t)(k0 + row) * D_ + c8);
        }
    };

    const int nTiles = (blockIdx.x + 1) * 4;     // keys up to q0+128 (causal)
    prefetch(0);
    for (int t = 0; t < nTiles; ++t) {
        const int k0 = t * 32;
        // ---- stage prefetched K (row-major) and V (transposed) into LDS ----
#pragma unroll
        for (int it = 0; it < 2; ++it) {
            int idx = tid + 256 * it;
            int row = idx >> 4;
            int c8  = (idx & 15) * 8;
            *(uint4*)(sK + row * 128 + c8) = kreg[it];
            const unsigned short* pv = (const unsigned short*)&vreg[it];
#pragma unroll
            for (int j2 = 0; j2 < 8; ++j2) sV[(c8 + j2) * 32 + row] = pv[j2];
        }
        __syncthreads();
        // prefetch next tile while this one is consumed by the matrix pipes
        if (t + 1 < nTiles) prefetch(k0 + 32);

        // uniform causal skip (keeps EXEC all-ones around WMMA)
        int act = __builtin_amdgcn_readfirstlane((k0 <= qw + 15) ? 1 : 0);
        if (act) {
            // ---- scores: Q(16x128) @ K_tile^T -> two 16x16 f32 frags ----
            v8f sc[2] = {};
#pragma unroll
            for (int j = 0; j < 2; ++j)
#pragma unroll
                for (int s4 = 0; s4 < 4; ++s4) {
                    int row = j * 16 + laneRow;
                    int c   = s4 * 32 + g8;
                    v16bf bfg = concatbf(*(const v8bf*)(sK + row * 128 + c),
                                         *(const v8bf*)(sK + row * 128 + c + 16));
                    sc[j] = wmma_bf16(qf[s4], bfg, sc[j]);
                }
            // ---- scale + causal mask ----
#pragma unroll
            for (int j = 0; j < 2; ++j) {
                int key = k0 + j * 16 + laneRow;
#pragma unroll
                for (int r = 0; r < 8; ++r) {
                    int qr = qw + r + hiOff;
                    float sv = sc[j][r] * SCALE_;
                    sc[j][r] = (key > qr) ? -3.0e38f : sv;
                }
            }
            // ---- online softmax (row = VGPR index within a lane half) ----
            float alpha[8];
#pragma unroll
            for (int r = 0; r < 8; ++r) {
                float mx = fmaxf(sc[0][r], sc[1][r]);
#pragma unroll
                for (int msk = 1; msk < 16; msk <<= 1)
                    mx = fmaxf(mx, __shfl_xor(mx, msk, 16));
                float mnew = fmaxf(mi[r], mx);
                alpha[r] = __expf(mi[r] - mnew);
                mi[r] = mnew;
                float p0 = __expf(sc[0][r] - mnew);
                float p1 = __expf(sc[1][r] - mnew);
                sc[0][r] = p0; sc[1][r] = p1;
                float rs = p0 + p1;
#pragma unroll
                for (int msk = 1; msk < 16; msk <<= 1)
                    rs += __shfl_xor(rs, msk, 16);
                li[r] = li[r] * alpha[r] + rs;
            }
#pragma unroll
            for (int t8 = 0; t8 < 8; ++t8)
#pragma unroll
                for (int r = 0; r < 8; ++r)
                    oacc[t8][r] *= alpha[r];

            // ---- P (C-layout) -> LDS -> A-fragment (same-wave, DS in-order) ----
            unsigned short* pw = sP + wave * 512;
#pragma unroll
            for (int j = 0; j < 2; ++j)
#pragma unroll
                for (int r = 0; r < 8; ++r)
                    pw[(r + hiOff) * 32 + j * 16 + laneRow] = f2bf(sc[j][r]);
            v16bf pf = concatbf(*(const v8bf*)(pw + laneRow * 32 + g8),
                                *(const v8bf*)(pw + laneRow * 32 + g8 + 16));
            // ---- O += P(16x32) @ V(32x128) ----
#pragma unroll
            for (int t8 = 0; t8 < 8; ++t8) {
                int row = t8 * 16 + laneRow;     // d-row in transposed sV
                v16bf vf = concatbf(*(const v8bf*)(sV + row * 32 + g8),
                                    *(const v8bf*)(sV + row * 32 + g8 + 16));
                oacc[t8] = wmma_bf16(pf, vf, oacc[t8]);
            }
        }
        __syncthreads();
    }

    // ---- normalize + write (b, s, h*D) bf16 ----
    const int b = bh / H_;
    const int h = bh % H_;
#pragma unroll
    for (int r = 0; r < 8; ++r) {
        float inv = 1.0f / li[r];
        int srow = qw + r + hiOff;
        size_t orow = (size_t)(b * S_ + srow) * HID_ + h * D_;
#pragma unroll
        for (int t8 = 0; t8 < 8; ++t8)
            Ob[orow + t8 * 16 + laneRow] = f2bf(oacc[t8][r] * inv);
    }
}

// ---------------------------------------------------------------------------
// Host launcher
// ---------------------------------------------------------------------------
extern "C" void kernel_launch(void* const* d_in, const int* in_sizes, int n_in,
                              void* d_out, int out_size, void* d_ws, size_t ws_size,
                              hipStream_t stream) {
    (void)in_sizes; (void)n_in; (void)out_size; (void)ws_size;
    const float* x     = (const float*)d_in[0];   // (B,S,HID)
    const float* w_qkv = (const float*)d_in[1];   // (3*HID, HID)
    const float* w_o   = (const float*)d_in[2];   // (HID, HID)
    float* out = (float*)d_out;                   // (B,S,HID)

    char* ws = (char*)d_ws;
    unsigned short* qkv = (unsigned short*)ws;                             // 4096*6144 bf16
    unsigned short* q   = (unsigned short*)(ws + (size_t)M_ * 3 * HID_ * 2);
    unsigned short* k   = q + (size_t)B_ * H_ * S_ * D_;
    unsigned short* v   = k + (size_t)B_ * H_ * S_ * D_;
    unsigned short* ao  = qkv;  // attention output reuses dead qkv region

    dim3 blk(256);
    gemm_xwt<float, float, unsigned short>
        <<<dim3((3 * HID_) / 128, M_ / 128), blk, 0, stream>>>(x, w_qkv, qkv, M_, 3 * HID_, HID_);
    rope_split<<<dim3((B_ * H_ * S_ * (D_ / 2)) / 256), blk, 0, stream>>>(qkv, q, k, v);
    flash_attn<<<dim3(S_ / 128, B_ * H_), blk, 0, stream>>>(q, k, v, ao);
    gemm_xwt<unsigned short, float, float>
        <<<dim3(HID_ / 128, M_ / 128), blk, 0, stream>>>(ao, w_o, out, M_, HID_, HID_);
}
